// DualRelationsPropagation_46935402610852
// MI455X (gfx1250) — matrix operand order repaired and checked by reading, not compiled, not run on hardware
//
#include <hip/hip_runtime.h>
#include <hip/hip_bf16.h>

#define HDIM 768
#define RREL 16
#define NSUP 64
#define NQRY 128
#define NTOT 192
#define NSTEPS 3

// GEMM macro-tile
#define BM 64
#define BN 128
#define BK 64

typedef __attribute__((ext_vector_type(16))) __bf16 v16bf;
typedef __attribute__((ext_vector_type(8)))  float  v8f;
typedef __attribute__((ext_vector_type(8)))  unsigned short v8us;

union FragU { v16bf f; v8us h[2]; };

#if defined(__AMDGCN__) && __has_builtin(__builtin_amdgcn_tensor_load_to_lds)
#define HAVE_TDM 1
#else
#define HAVE_TDM 0
#endif

__device__ __forceinline__ unsigned short f2bf_us(float f) {
  unsigned u = __builtin_bit_cast(unsigned, f);
  unsigned r = u + 0x7fffu + ((u >> 16) & 1u);   // round-to-nearest-even
  return (unsigned short)(r >> 16);
}
__device__ __forceinline__ __bf16 f2bf(float f) {
  unsigned short s = f2bf_us(f);
  return __builtin_bit_cast(__bf16, s);
}
__device__ __forceinline__ v8f wmma_bf16x32(FragU a, FragU b, v8f c) {
  return __builtin_amdgcn_wmma_f32_16x16x32_bf16(false, a.f, false, b.f,
                                                 (short)0, c, false, false);
}

#if HAVE_TDM
typedef unsigned int u32x4 __attribute__((ext_vector_type(4)));
typedef int          i32x8 __attribute__((ext_vector_type(8)));
typedef int          i32x4 __attribute__((ext_vector_type(4)));

// Issue one TDM 2D tile load (bf16 elements): tile_rows x tile_k from a
// row-major matrix with leading dim `ld` (halves) into LDS at lds_off.
// D# packing per CDNA5 ISA 8.3/8.4 (group0 128b, group1 256b), groups 2/3 zero.
// This toolchain exposes the 6-arg builtin:
//   (u32x4 g0, i32x8 g1, i32x4 g2, i32x4 g3, i32x8 extra, i32 cpol)
__device__ __forceinline__ void tdm_load_2d(const void* gptr, unsigned lds_off,
                                            unsigned tile_k, unsigned tile_rows,
                                            unsigned ld, unsigned rows_rem,
                                            unsigned k_rem) {
  unsigned long long ga = (unsigned long long)gptr;
  u32x4 g0;
  g0[0] = 1u;                                           // count=1, user D#
  g0[1] = lds_off;                                      // lds_addr (bytes)
  g0[2] = (unsigned)ga;                                 // global_addr[31:0]
  g0[3] = ((unsigned)(ga >> 32) & 0x01FFFFFFu) | (2u << 30); // addr[56:32]|type=2
  unsigned td0 = k_rem;                                 // tensor_dim0 (halves)
  unsigned td1 = rows_rem;                              // tensor_dim1
  unsigned long long s0 = ld;                           // dim0 stride (halves)
  unsigned long long s1 = (unsigned long long)ld * rows_rem;
  i32x8 g1;
  g1[0] = (int)(1u << 16);                              // data_size = 2 bytes
  g1[1] = (int)((td0 & 0xFFFFu) << 16);                 // td0[15:0]
  g1[2] = (int)(((td0 >> 16) & 0xFFFFu) | ((td1 & 0xFFFFu) << 16));
  g1[3] = (int)(((td1 >> 16) & 0xFFFFu) | ((tile_k & 0xFFFFu) << 16)); // tile_dim0
  g1[4] = (int)(tile_rows & 0xFFFFu);                   // tile_dim1, tile_dim2=0
  g1[5] = (int)(unsigned)(s0 & 0xFFFFFFFFu);            // stride0[31:0]
  g1[6] = (int)(((unsigned)((s0 >> 32) & 0xFFFFu)) | (((unsigned)(s1 & 0xFFFFu)) << 16));
  g1[7] = (int)(unsigned)((s1 >> 16) & 0xFFFFFFFFu);    // stride1[47:16]
  i32x4 zz4 = {0, 0, 0, 0};
  i32x8 zz8 = {0, 0, 0, 0, 0, 0, 0, 0};
  __builtin_amdgcn_tensor_load_to_lds(g0, g1, zz4, zz4, zz8, 0);
}
__device__ __forceinline__ void tdm_wait0() {
#if __has_builtin(__builtin_amdgcn_s_wait_tensorcnt)
  __builtin_amdgcn_s_wait_tensorcnt(0);
#else
  asm volatile("s_wait_tensorcnt 0" ::: "memory");
#endif
}
#endif  // HAVE_TDM

// ---------------------------------------------------------------------------
// Elementwise f32 -> bf16 convert
// ---------------------------------------------------------------------------
__global__ __launch_bounds__(256)
void cvt_bf16(const float* __restrict__ in, unsigned short* __restrict__ out, int n) {
  int i = blockIdx.x * 256 + threadIdx.x;
  if (i < n) out[i] = f2bf_us(in[i]);
}

// ---------------------------------------------------------------------------
// Batched transpose+convert: out[z][d][h] = bf16(in[z][h][d]), H x H per batch
// ---------------------------------------------------------------------------
__global__ __launch_bounds__(256)
void transpose_cvt(const float* __restrict__ in, unsigned short* __restrict__ out) {
  long long base = (long long)blockIdx.z * HDIM * HDIM;
  int i = blockIdx.x * 256 + threadIdx.x;          // output-linear index
  if (i < HDIM * HDIM) {
    int d = i / HDIM, hh = i - d * HDIM;
    out[base + i] = f2bf_us(in[base + (long long)hh * HDIM + d]);
  }
}

// ---------------------------------------------------------------------------
// bf16 GEMM: C[M,N] (f32, opt bf16 mirror, opt bias) = A[M,K] @ B[N,K]^T.
// Workgroup 256 threads = 8 waves; macro-tile 64Mx128Nx64K; wave tile 32x32
// (4 accumulators -> 8 v_wmma per K-slab). A/B staged to LDS by the Tensor
// Data Mover (TENSOR_LOAD_TO_LDS + s_wait_tensorcnt) when available, else by
// cooperative b128 loads. Batched over blockIdx.z.
// ---------------------------------------------------------------------------
__global__ __launch_bounds__(256)
void gemm_bf16_nt(const unsigned short* __restrict__ A, int lda, long long sA,
                  const unsigned short* __restrict__ B, int ldb, long long sB,
                  float* __restrict__ C, int ldc, long long sC,
                  unsigned short* __restrict__ C2, int ldc2, long long sC2,
                  const float* __restrict__ bias, int M, int Nrows, int K) {
  __shared__ unsigned short Asm[BM * BK];
  __shared__ unsigned short Bsm[BN * BK];
  const int z = blockIdx.z;
  A += (long long)z * sA;
  B += (long long)z * sB;
  C += (long long)z * sC;
  unsigned short* C2p = C2 ? (C2 + (long long)z * sC2) : nullptr;

  const int m0 = blockIdx.x * BM, n0 = blockIdx.y * BN;
  const int tid = threadIdx.x, wave = tid >> 5, lane = tid & 31;
  const int wm = (wave & 1) * 32;        // wave M offset in tile
  const int wn = (wave >> 1) * 32;       // wave N offset in tile
  const int nl = lane & 15;
  // CDNA5 16-bit fragment lane maps:
  //  A 16x32: lane<16 -> K{0..7,16..23}, lane>=16 -> K{8..15,24..31}; row=lane%16
  //  B 32x16: lane<16 -> K{0..7,16..23}, lane>=16 -> K{8..15,24..31}; col=lane%16
  const int aoff = (lane < 16) ? 0 : 8;
  const int koff = (lane < 16) ? 0 : 16;

  v8f acc00 = {}, acc01 = {}, acc10 = {}, acc11 = {};

  for (int kb = 0; kb < K; kb += BK) {
#if HAVE_TDM
    if (wave == 0) {
      tdm_load_2d(&A[(long long)m0 * lda + kb], (unsigned)(unsigned long long)(size_t)Asm,
                  BK, BM, (unsigned)lda, (unsigned)(M - m0), (unsigned)(K - kb));
      tdm_load_2d(&B[(long long)n0 * ldb + kb], (unsigned)(unsigned long long)(size_t)Bsm,
                  BK, BN, (unsigned)ldb, (unsigned)(Nrows - n0), (unsigned)(K - kb));
      tdm_wait0();
    }
#else
    // A: 64x64 halves = 512 x 16B chunks (2/thread); B: 128x64 = 1024 (4/thread)
#pragma unroll
    for (int q = 0; q < 2; ++q) {
      int c = tid * 2 + q;
      int row = c >> 3, o = (c & 7) * 8;
      *(v8us*)&Asm[row * BK + o] = *(const v8us*)&A[(long long)(m0 + row) * lda + kb + o];
    }
#pragma unroll
    for (int q = 0; q < 4; ++q) {
      int c = tid * 4 + q;
      int row = c >> 3, o = (c & 7) * 8;
      *(v8us*)&Bsm[row * BK + o] = *(const v8us*)&B[(long long)(n0 + row) * ldb + kb + o];
    }
    if (kb + BK < K) {
      __builtin_prefetch(&A[(long long)(m0 + (tid >> 3)) * lda + kb + BK], 0, 1);
      __builtin_prefetch(&B[(long long)(n0 + (tid >> 1)) * ldb + kb + BK], 0, 1);
    }
#endif
    __syncthreads();
#pragma unroll
    for (int ks = 0; ks < BK; ks += 32) {
      FragU a0, a1, b0, b1;
      const int ar = (wm + nl) * BK + ks;
      a0.h[0] = *(const v8us*)&Asm[ar + aoff];
      a0.h[1] = *(const v8us*)&Asm[ar + aoff + 16];
      a1.h[0] = *(const v8us*)&Asm[ar + 16 * BK + aoff];
      a1.h[1] = *(const v8us*)&Asm[ar + 16 * BK + aoff + 16];
      const int br = (wn + nl) * BK + ks;
      b0.h[0] = *(const v8us*)&Bsm[br + aoff];
      b0.h[1] = *(const v8us*)&Bsm[br + aoff + 16];
      b1.h[0] = *(const v8us*)&Bsm[br + 16 * BK + aoff];
      b1.h[1] = *(const v8us*)&Bsm[br + 16 * BK + aoff + 16];
      acc00 = wmma_bf16x32(a0, b0, acc00);
      acc01 = wmma_bf16x32(a0, b1, acc01);
      acc10 = wmma_bf16x32(a1, b0, acc10);
      acc11 = wmma_bf16x32(a1, b1, acc11);
    }
    __syncthreads();
  }

  const int col0 = n0 + wn + nl, col1 = col0 + 16;
  const float b0v = bias ? bias[col0] : 0.0f;
  const float b1v = bias ? bias[col1] : 0.0f;
  const int mh = (lane < 16) ? 0 : 8;
#pragma unroll
  for (int v = 0; v < 8; ++v) {
    const long long m = m0 + wm + v + mh;
    float x00 = acc00[v] + b0v, x01 = acc01[v] + b1v;
    float x10 = acc10[v] + b0v, x11 = acc11[v] + b1v;
    C[m * ldc + col0] = x00;
    C[m * ldc + col1] = x01;
    C[(m + 16) * ldc + col0] = x10;
    C[(m + 16) * ldc + col1] = x11;
    if (C2p) {
      C2p[m * ldc2 + col0] = f2bf_us(x00);
      C2p[m * ldc2 + col1] = f2bf_us(x01);
      C2p[(m + 16) * ldc2 + col0] = f2bf_us(x10);
      C2p[(m + 16) * ldc2 + col1] = f2bf_us(x11);
    }
  }
}

// ---------------------------------------------------------------------------
// Pair relation net: hid = relu(Arow[i]+Brow[j]) built in registers (bf16 A
// fragment), WMMA against W2 (R=16 outputs), softmax over R via shfl_xor.
// Block = (i-tile 16) x (j-tile 16); wave handles 2 groups of 16 pairs.
// ---------------------------------------------------------------------------
__global__ __launch_bounds__(256)
void pair_relnet(const float* __restrict__ Arow, const float* __restrict__ Brow,
                 const unsigned short* __restrict__ W2t, const float* __restrict__ b2,
                 unsigned short* __restrict__ swdw, int slot) {
  const int it = blockIdx.x * 16, jt = blockIdx.y * 16;
  const int wave = threadIdx.x >> 5, lane = threadIdx.x & 31;
  const int tl = lane & 15;                 // pair row in group == local j
  const int j = jt + tl;
  const int k0 = (lane < 16) ? 0 : 8;
  const int koff = (lane < 16) ? 0 : 16;
  const int r = lane & 15;
  for (int gg = 0; gg < 2; ++gg) {
    const int i = it + wave * 2 + gg;
    v8f acc = {};
    for (int kb = 0; kb < HDIM; kb += 32) {
      v8f a0 = *(const v8f*)&Arow[(long long)i * HDIM + kb + k0];
      v8f a1 = *(const v8f*)&Arow[(long long)i * HDIM + kb + k0 + 16];
      v8f c0 = *(const v8f*)&Brow[(long long)j * HDIM + kb + k0];
      v8f c1 = *(const v8f*)&Brow[(long long)j * HDIM + kb + k0 + 16];
      FragU af;
#pragma unroll
      for (int e = 0; e < 8; ++e) { float v = a0[e] + c0[e]; af.f[e] = f2bf(v > 0.f ? v : 0.f); }
#pragma unroll
      for (int e = 0; e < 8; ++e) { float v = a1[e] + c1[e]; af.f[8 + e] = f2bf(v > 0.f ? v : 0.f); }
      FragU bf_;
      bf_.h[0] = *(const v8us*)&W2t[(long long)tl * HDIM + kb + k0];
      bf_.h[1] = *(const v8us*)&W2t[(long long)tl * HDIM + kb + k0 + 16];
      acc = wmma_bf16x32(af, bf_, acc);
    }
    const float bb = b2[r];
#pragma unroll
    for (int v = 0; v < 8; ++v) {
      float val = acc[v] + bb;
      float mx = val;
      for (int mk = 1; mk < 16; mk <<= 1) mx = fmaxf(mx, __shfl_xor(mx, mk, 32));
      float e = __expf(val - mx);
      float s = e;
      for (int mk = 1; mk < 16; mk <<= 1) s += __shfl_xor(s, mk, 32);
      const int pm = v + ((lane < 16) ? 0 : 8);
      const int jj = jt + pm;
      swdw[((long long)i * NTOT + jj) * 32 + slot + r] = f2bf_us(e / s);
    }
  }
}

// ---------------------------------------------------------------------------
// Repack tsW f32 [N*R, H] -> TT bf16 [N][H][32] at slot offset (K-major rows
// so the message WMMA B operand is contiguous per lane).
// ---------------------------------------------------------------------------
__global__ __launch_bounds__(256)
void repack_tt(const float* __restrict__ tsW, unsigned short* __restrict__ TT, int slot) {
  int idx = blockIdx.x * 256 + threadIdx.x;          // over N*R*H
  if (idx < NTOT * RREL * HDIM) {
    int row = idx / HDIM, d = idx - row * HDIM;      // row = j*16 + r
    int j = row >> 4, rr = row & 15;
    TT[((long long)j * HDIM + d) * 32 + slot + rr] = f2bf_us(tsW[idx]);
  }
}

// ---------------------------------------------------------------------------
// Message accumulate: per j one WMMA  swdw[16 i, 32] x T_j[32, 16 d],
// +bm1, relu, running sum over j; write mean as bf16 x_pre.
// Grid (12 i-tiles, 6); wave -> one 16-d tile (8 waves x 6 = 48 d-tiles).
// ---------------------------------------------------------------------------
__global__ __launch_bounds__(256)
void msg_accum(const unsigned short* __restrict__ swdw,
               const unsigned short* __restrict__ TT,
               const float* __restrict__ bm1,
               unsigned short* __restrict__ xpre) {
  const int it = blockIdx.x * 16;
  const int wave = threadIdx.x >> 5, lane = threadIdx.x & 31;
  const int d0 = (blockIdx.y * 8 + wave) * 16;
  const int nl = lane & 15;
  const int aoff = (lane < 16) ? 0 : 8;
  const int koff = (lane < 16) ? 0 : 16;
  const float bb = bm1[d0 + nl];
  v8f acc = {};
  const v8f zero = {};
  for (int j = 0; j < NTOT; ++j) {
    const unsigned short* ap = swdw + ((long long)(it + nl) * NTOT + j) * 32;
    FragU a;
    a.h[0] = *(const v8us*)&ap[aoff];
    a.h[1] = *(const v8us*)&ap[aoff + 16];
    const unsigned short* bp = TT + ((long long)j * HDIM + d0 + nl) * 32 + koff;
    FragU b;
    b.h[0] = *(const v8us*)&bp[aoff - koff];
    b.h[1] = *(const v8us*)&bp[aoff - koff + 16];
    v8f t = wmma_bf16x32(a, b, zero);
#pragma unroll
    for (int v = 0; v < 8; ++v) { float q = t[v] + bb; acc[v] += (q > 0.f ? q : 0.f); }
  }
  const float inv = 1.0f / (float)NTOT;
  const int mh = (lane < 16) ? 0 : 8;
#pragma unroll
  for (int v = 0; v < 8; ++v) {
    const int i = it + v + mh;
    xpre[(long long)i * HDIM + d0 + nl] = f2bf_us(acc[v] * inv);
  }
}

// ---------------------------------------------------------------------------
// GRU cell update (torch gate order r, z, n); biases folded into gx/gh GEMMs.
// ---------------------------------------------------------------------------
__global__ __launch_bounds__(256)
void gru_update(const float* __restrict__ gx, const float* __restrict__ gh,
                float* __restrict__ h, unsigned short* __restrict__ hbf) {
  int idx = blockIdx.x * 256 + threadIdx.x;          // over N*H
  if (idx < NTOT * HDIM) {
    int i = idx / HDIM, c = idx - i * HDIM;
    const float* gxr = gx + (long long)i * (3 * HDIM);
    const float* ghr = gh + (long long)i * (3 * HDIM);
    float r = 1.f / (1.f + __expf(-(gxr[c] + ghr[c])));
    float z = 1.f / (1.f + __expf(-(gxr[HDIM + c] + ghr[HDIM + c])));
    float n = tanhf(gxr[2 * HDIM + c] + r * ghr[2 * HDIM + c]);
    float hn = (1.f - z) * n + z * h[idx];
    h[idx] = hn;
    hbf[idx] = f2bf_us(hn);
  }
}

// ---------------------------------------------------------------------------
// Host orchestration
// ---------------------------------------------------------------------------
extern "C" void kernel_launch(void* const* d_in, const int* in_sizes, int n_in,
                              void* d_out, int out_size, void* d_ws, size_t ws_size,
                              hipStream_t stream) {
  (void)in_sizes; (void)n_in; (void)out_size; (void)ws_size;
  const float* aspect = (const float*)d_in[0];
  const float* query  = (const float*)d_in[1];
  /* d_in[2] support_labels: unused by the reference forward */
  const float* W1s = (const float*)d_in[3];
  const float* b1s = (const float*)d_in[4];
  const float* W2s = (const float*)d_in[5];
  const float* b2s = (const float*)d_in[6];
  const float* W1d = (const float*)d_in[7];
  const float* b1d = (const float*)d_in[8];
  const float* W2d = (const float*)d_in[9];
  const float* b2d = (const float*)d_in[10];
  const float* Rs  = (const float*)d_in[11];
  const float* Rd  = (const float*)d_in[12];
  const float* Wm1 = (const float*)d_in[13];
  const float* bm1 = (const float*)d_in[14];
  const float* Wm2 = (const float*)d_in[15];
  const float* bm2 = (const float*)d_in[16];
  const float* Wih = (const float*)d_in[17];
  const float* Whh = (const float*)d_in[18];
  const float* bih = (const float*)d_in[19];
  const float* bhh = (const float*)d_in[20];

  const int H = HDIM, N = NTOT, R = RREL;
  size_t off = 0;
  auto take = [&](size_t bytes) -> size_t {
    off = (off + 255) & ~(size_t)255;
    size_t o = off; off += bytes; return o;
  };
  char* ws = (char*)d_ws;
  float*          h_f    = (float*)(ws + take((size_t)N * H * 4));
  unsigned short* h_bf   = (unsigned short*)(ws + take((size_t)N * H * 2));
  unsigned short* W1s_bf = (unsigned short*)(ws + take((size_t)H * 2 * H * 2));
  unsigned short* W1d_bf = (unsigned short*)(ws + take((size_t)H * 2 * H * 2));
  unsigned short* W2s_bf = (unsigned short*)(ws + take((size_t)R * H * 2));
  unsigned short* W2d_bf = (unsigned short*)(ws + take((size_t)R * H * 2));
  unsigned short* Wm1_bf = (unsigned short*)(ws + take((size_t)H * H * 2));
  unsigned short* Wm2_bf = (unsigned short*)(ws + take((size_t)H * H * 2));
  unsigned short* Wih_bf = (unsigned short*)(ws + take((size_t)3 * H * H * 2));
  unsigned short* Whh_bf = (unsigned short*)(ws + take((size_t)3 * H * H * 2));
  unsigned short* RsT_bf = (unsigned short*)(ws + take((size_t)R * H * H * 2));
  unsigned short* RdT_bf = (unsigned short*)(ws + take((size_t)R * H * H * 2));
  float*          As_s   = (float*)(ws + take((size_t)N * H * 4));
  float*          Bs_s   = (float*)(ws + take((size_t)N * H * 4));
  float*          As_d   = (float*)(ws + take((size_t)N * H * 4));
  float*          Bs_d   = (float*)(ws + take((size_t)N * H * 4));
  unsigned short* swdw   = (unsigned short*)(ws + take((size_t)N * N * 32 * 2));
  float*          ts_f   = (float*)(ws + take((size_t)N * R * H * 4));
  unsigned short* ts_bf  = (unsigned short*)(ws + take((size_t)N * R * H * 2));
  float*          tsW_f  = (float*)(ws + take((size_t)N * R * H * 4));
  unsigned short* TT     = (unsigned short*)(ws + take((size_t)N * H * 32 * 2));
  unsigned short* xp_bf  = (unsigned short*)(ws + take((size_t)N * H * 2));
  float*          x_f    = (float*)(ws + take((size_t)N * H * 4));
  unsigned short* x_bf   = (unsigned short*)(ws + take((size_t)N * H * 2));
  float*          gx     = (float*)(ws + take((size_t)N * 3 * H * 4));
  float*          gh     = (float*)(ws + take((size_t)N * 3 * H * 4));

  auto cvt = [&](const float* src, unsigned short* dst, int n) {
    cvt_bf16<<<dim3((n + 255) / 256), dim3(256), 0, stream>>>(src, dst, n);
  };
  auto gemm = [&](const unsigned short* A, int lda, long long sA,
                  const unsigned short* B, int ldb, long long sB,
                  float* C, int ldc, long long sC,
                  unsigned short* C2, int ldc2, long long sC2,
                  const float* bias, int M, int Nc, int K, int batch) {
    gemm_bf16_nt<<<dim3(M / BM, Nc / BN, batch), dim3(256), 0, stream>>>(
        A, lda, sA, B, ldb, sB, C, ldc, sC, C2, ldc2, sC2, bias, M, Nc, K);
  };

  // ---- one-time prep ----
  hipMemcpyAsync(h_f, aspect, (size_t)NSUP * H * 4, hipMemcpyDeviceToDevice, stream);
  hipMemcpyAsync(h_f + (size_t)NSUP * H, query, (size_t)NQRY * H * 4,
                 hipMemcpyDeviceToDevice, stream);
  cvt(h_f, h_bf, N * H);
  cvt(W1s, W1s_bf, H * 2 * H);
  cvt(W1d, W1d_bf, H * 2 * H);
  cvt(W2s, W2s_bf, R * H);
  cvt(W2d, W2d_bf, R * H);
  cvt(Wm1, Wm1_bf, H * H);
  cvt(Wm2, Wm2_bf, H * H);
  cvt(Wih, Wih_bf, 3 * H * H);
  cvt(Whh, Whh_bf, 3 * H * H);
  transpose_cvt<<<dim3((H * H) / 256, 1, R), dim3(256), 0, stream>>>(Rs, RsT_bf);
  transpose_cvt<<<dim3((H * H) / 256, 1, R), dim3(256), 0, stream>>>(Rd, RdT_bf);

  // ---- propagation steps ----
  for (int step = 0; step < NSTEPS; ++step) {
    // Relation-net linear halves (b1 folded into the "i" half)
    gemm(h_bf, H, 0, W1s_bf, 2 * H, 0, As_s, H, 0, nullptr, 0, 0, b1s, N, H, H, 1);
    gemm(h_bf, H, 0, W1s_bf + H, 2 * H, 0, Bs_s, H, 0, nullptr, 0, 0, nullptr, N, H, H, 1);
    gemm(h_bf, H, 0, W1d_bf, 2 * H, 0, As_d, H, 0, nullptr, 0, 0, b1d, N, H, H, 1);
    gemm(h_bf, H, 0, W1d_bf + H, 2 * H, 0, Bs_d, H, 0, nullptr, 0, 0, nullptr, N, H, H, 1);

    // Pairwise relation weights (softmax over R) -> swdw slots [0..15]/[16..31]
    pair_relnet<<<dim3(N / 16, N / 16), dim3(256), 0, stream>>>(As_s, Bs_s, W2s_bf, b2s, swdw, 0);
    pair_relnet<<<dim3(N / 16, N / 16), dim3(256), 0, stream>>>(As_d, Bs_d, W2d_bf, b2d, swdw, 16);

    // ts = h @ Rs[r]  (batched over r), then tsW = ts @ Wm1^T, pack into TT
    gemm(h_bf, H, 0, RsT_bf, H, (long long)H * H, ts_f, R * H, H, ts_bf, R * H, H,
         nullptr, N, H, H, R);
    gemm(ts_bf, H, 0, Wm1_bf, H, 0, tsW_f, H, 0, nullptr, 0, 0, nullptr, N * R, H, H, 1);
    repack_tt<<<dim3((N * R * H) / 256), dim3(256), 0, stream>>>(tsW_f, TT, 0);

    gemm(h_bf, H, 0, RdT_bf, H, (long long)H * H, ts_f, R * H, H, ts_bf, R * H, H,
         nullptr, N, H, H, R);
    gemm(ts_bf, H, 0, Wm1_bf, H, 0, tsW_f, H, 0, nullptr, 0, 0, nullptr, N * R, H, H, 1);
    repack_tt<<<dim3((N * R * H) / 256), dim3(256), 0, stream>>>(tsW_f, TT, 16);

    // mean_j relu(msgs@Wm1^T + bm1)  -> x_pre (bf16)
    msg_accum<<<dim3(N / 16, 6), dim3(256), 0, stream>>>(swdw, TT, bm1, xp_bf);

    // x = x_pre @ Wm2^T + bm2 ; GRU gate GEMMs ; GRU update
    gemm(xp_bf, H, 0, Wm2_bf, H, 0, x_f, H, 0, x_bf, H, 0, bm2, N, H, H, 1);
    gemm(x_bf, H, 0, Wih_bf, H, 0, gx, 3 * H, 0, nullptr, 0, 0, bih, N, 3 * H, H, 1);
    gemm(h_bf, H, 0, Whh_bf, H, 0, gh, 3 * H, 0, nullptr, 0, 0, bhh, N, 3 * H, H, 1);
    gru_update<<<dim3((N * H) / 256), dim3(256), 0, stream>>>(gx, gh, h_f, h_bf);
  }

  // output: updated query embeddings h[NS:]
  hipMemcpyAsync(d_out, h_f + (size_t)NSUP * H, (size_t)NQRY * H * 4,
                 hipMemcpyDeviceToDevice, stream);
}
